// VarianceAdopter_83382495085095
// MI455X (gfx1250) — matrix operand order repaired and checked by reading, not compiled, not run on hardware
//
#include <hip/hip_runtime.h>
#include <hip/hip_bf16.h>

#define BB  32
#define CH  384
#define TXX 512
#define TYY 2048

typedef __bf16 bf16_t;
typedef __attribute__((ext_vector_type(16))) __bf16 v16bf;
typedef __attribute__((ext_vector_type(8)))  float  v8f;
typedef __attribute__((ext_vector_type(4)))  float  f32x4;

union V16frag { v16bf v; f32x4 q[2]; };

__device__ __forceinline__ bf16_t f2bf(float f) {
  unsigned u = __float_as_uint(f);
  unsigned r = (u + 0x7FFFu + ((u >> 16) & 1u)) >> 16;   // round-to-nearest-even
  unsigned short s = (unsigned short)r;
  return __builtin_bit_cast(bf16_t, s);
}

// ---------------------------------------------------------------------------
// bf16 WMMA GEMM:  D[b][m][n] = sum_tap sum_k A[b,tap][m][k] * BT[b][n+shift][k]
//   A   : bf16 row-major [M][Kdim] (batch/tap strides); BT : bf16 [N][Kdim]
//   epi : 0 = plain store, 1 = +bias, SiLU, *mask
// Block = 256 threads (8 waves). Tile 128(M) x 64(N); wave = 32x32 patch,
// 4 accumulators. Double-buffered LDS, register-staged pipeline, 1 barrier/iter.
// ---------------------------------------------------------------------------
__global__ __launch_bounds__(256) void wmma_gemm_kernel(
    const bf16_t* __restrict__ A, long a_bs, long a_ts, int ntaps,
    const bf16_t* __restrict__ BT, long b_bs,
    float* __restrict__ D, long d_bs,
    int N, int Kdim,
    const float* __restrict__ bias, const float* __restrict__ mask, int epi)
{
  __shared__ bf16_t As[2][128 * 40];   // stride 40 bf16 = 80B rows: 16B aligned, bank-friendly
  __shared__ bf16_t Bs[2][64 * 40];

  const int b    = blockIdx.z;
  const int m0   = blockIdx.y * 128;
  const int n0   = blockIdx.x * 64;
  const int tid  = threadIdx.x;
  const int lane = tid & 31;
  const int wave = tid >> 5;
  const int wm   = (wave & 3) * 32;     // wave M offset within tile (4 groups)
  const int wn   = (wave >> 2) * 32;    // wave N offset within tile (2 groups)

  const int tr = tid >> 2;              // 0..63 loader row
  const int tk = (tid & 3) * 8;         // 0,8,16,24 (8 bf16 = 16B chunk)

  const bf16_t* Bb = BT + (long)b * b_bs;
  const int kSteps = Kdim >> 5;
  const int iters  = ntaps * kSteps;

  v8f acc[4];
#pragma unroll
  for (int i = 0; i < 4; ++i) acc[i] = (v8f){0.f,0.f,0.f,0.f,0.f,0.f,0.f,0.f};

  // register-staged tile fetch for iteration `it`
  auto fetch = [&](int it, f32x4& a0, f32x4& a1, f32x4& bv) {
    const int tap = it / kSteps;
    const int kk  = (it - tap * kSteps) << 5;
    const bf16_t* Ab = A + (long)b * a_bs + (long)tap * a_ts;
    const bf16_t* ap = Ab + (long)(m0 + tr) * Kdim + kk + tk;
    a0 = *(const f32x4*)(ap);
    a1 = *(const f32x4*)(ap + (long)64 * Kdim);
    const int shift = tap - (ntaps >> 1);          // {-1,0,1} for K=3, 0 for K=1
    const int srcn  = n0 + tr + shift;
    bv = (f32x4){0.f, 0.f, 0.f, 0.f};
    if (srcn >= 0 && srcn < N) {
      const bf16_t* bp = Bb + (long)srcn * Kdim + kk + tk;
      bv = *(const f32x4*)(bp);
      __builtin_prefetch(bp + 32, 0, 1);           // next K-chunk of the B stream
    }
  };
  auto stash = [&](int buf, const f32x4& a0, const f32x4& a1, const f32x4& bv) {
    *(f32x4*)(&As[buf][tr * 40 + tk])        = a0;
    *(f32x4*)(&As[buf][(tr + 64) * 40 + tk]) = a1;
    *(f32x4*)(&Bs[buf][tr * 40 + tk])        = bv;
  };

  {
    f32x4 a0, a1, bv;
    fetch(0, a0, a1, bv);
    stash(0, a0, a1, bv);
  }
  __syncthreads();

  const int r  = lane & 15;
  const int hf = lane >> 4;

  for (int it = 0; it < iters; ++it) {
    const int cur = it & 1, nxt = cur ^ 1;
    f32x4 na0, na1, nb;
    const bool more = (it + 1) < iters;
    if (more) fetch(it + 1, na0, na1, nb);

    // fragments per ISA wave32 VGPR layouts
    V16frag a0f, a1f, b0f, b1f;
    {
      const bf16_t* ap0 = &As[cur][(wm + r) * 40 + hf * 8];
      a0f.q[0] = *(const f32x4*)(ap0);        // K = hf*8 .. +8
      a0f.q[1] = *(const f32x4*)(ap0 + 16);   // K = hf*8+16 .. +24
      const bf16_t* ap1 = &As[cur][(wm + 16 + r) * 40 + hf * 8];
      a1f.q[0] = *(const f32x4*)(ap1);
      a1f.q[1] = *(const f32x4*)(ap1 + 16);
      const bf16_t* bp0 = &Bs[cur][(wn + r) * 40 + hf * 8];
      b0f.q[0] = *(const f32x4*)(bp0);        // col = r, K = hf*8 .. +8, hf*8+16 .. +24
      b0f.q[1] = *(const f32x4*)(bp0 + 16);
      const bf16_t* bp1 = &Bs[cur][(wn + 16 + r) * 40 + hf * 8];
      b1f.q[0] = *(const f32x4*)(bp1);
      b1f.q[1] = *(const f32x4*)(bp1 + 16);
    }
    acc[0] = __builtin_amdgcn_wmma_f32_16x16x32_bf16(false, a0f.v, false, b0f.v,
                                                     (short)0, acc[0], false, false);
    acc[1] = __builtin_amdgcn_wmma_f32_16x16x32_bf16(false, a0f.v, false, b1f.v,
                                                     (short)0, acc[1], false, false);
    acc[2] = __builtin_amdgcn_wmma_f32_16x16x32_bf16(false, a1f.v, false, b0f.v,
                                                     (short)0, acc[2], false, false);
    acc[3] = __builtin_amdgcn_wmma_f32_16x16x32_bf16(false, a1f.v, false, b1f.v,
                                                     (short)0, acc[3], false, false);

    if (more) stash(nxt, na0, na1, nb);
    __syncthreads();
  }

  // epilogue: D(m,n): lane = n + (m>=8-half)*16, element e -> M = e + 8*hf
  float* Dp = D + (long)b * d_bs;
#pragma unroll
  for (int im = 0; im < 2; ++im) {
#pragma unroll
    for (int in_ = 0; in_ < 2; ++in_) {
      v8f a = acc[im * 2 + in_];
      const int gn = n0 + wn + in_ * 16 + r;
      float mk = 1.f;
      if (epi == 1) mk = mask[(long)b * N + gn];
#pragma unroll
      for (int e = 0; e < 8; ++e) {
        const int gm = m0 + wm + im * 16 + hf * 8 + e;
        float v = a[e];
        if (epi == 1) {
          v += bias[gm];
          v = v / (1.f + __expf(-v));   // SiLU
          v *= mk;
        }
        Dp[(long)gm * N + gn] = v;
      }
    }
  }
}

// ---------------------------------------------------------------------------
// Channel LayerNorm over C=384 of X[B][C][T] -> bf16 transposed Y[B][T][C]
// ---------------------------------------------------------------------------
__global__ __launch_bounds__(256) void ln_to_bf16T_kernel(
    const float* __restrict__ X, const float* __restrict__ g,
    const float* __restrict__ bta, bf16_t* __restrict__ Y, int T)
{
  const int b  = blockIdx.y;
  const int t0 = blockIdx.x * 32;
  const int tl = threadIdx.x & 31;
  const int cg = threadIdx.x >> 5;            // 0..7
  const float* xp = X + (long)b * CH * T + t0 + tl;

  float s = 0.f, sq = 0.f;
  for (int j = 0; j < 48; ++j) {
    float v = xp[(long)(cg * 48 + j) * T];
    s += v; sq += v * v;
  }
  __shared__ float ssum[8][32], ssq[8][32], smean[32], srstd[32];
  ssum[cg][tl] = s; ssq[cg][tl] = sq;
  __syncthreads();
  if (threadIdx.x < 32) {
    float a = 0.f, q = 0.f;
    for (int i = 0; i < 8; ++i) { a += ssum[i][tl]; q += ssq[i][tl]; }
    float mean = a * (1.f / CH);
    float var  = q * (1.f / CH) - mean * mean;
    smean[tl] = mean;
    srstd[tl] = rsqrtf(var + 1e-5f);
  }
  __syncthreads();
  const float mean = smean[tl], rstd = srstd[tl];
  bf16_t* yp = Y + ((long)b * T + t0 + tl) * CH;
  for (int j = 0; j < 48; ++j) {
    const int c = cg * 48 + j;
    float v = xp[(long)c * T];
    yp[c] = f2bf((v - mean) * rstd * g[c] + bta[c]);
  }
}

// Final 1x1 conv to one channel: out[b][t] = (sum_c X[b][c][t]*w[c] + ob) * mask
__global__ __launch_bounds__(256) void conv1x1_kernel(
    const float* __restrict__ X, const float* __restrict__ w,
    const float* __restrict__ ob, const float* __restrict__ mask,
    float* __restrict__ out, int T)
{
  const int b  = blockIdx.y;
  const int t0 = blockIdx.x * 32;
  const int tl = threadIdx.x & 31;
  const int cg = threadIdx.x >> 5;
  const float* xp = X + (long)b * CH * T + t0 + tl;
  float s = 0.f;
  for (int j = 0; j < 48; ++j) {
    const int c = cg * 48 + j;
    s += xp[(long)c * T] * w[c];
  }
  __shared__ float ssum[8][32];
  ssum[cg][tl] = s;
  __syncthreads();
  if (threadIdx.x < 32) {
    float a = 0.f;
    for (int i = 0; i < 8; ++i) a += ssum[i][tl];
    out[(long)b * T + t0 + tl] = (a + ob[0]) * mask[(long)b * T + t0 + tl];
  }
}

// conv weights [NL][O][I][K] f32 -> [NL][K][O][I] bf16
__global__ void wprep_kernel(const float* __restrict__ cw, bf16_t* __restrict__ out)
{
  const long CC2 = (long)CH * CH;
  long i = (long)blockIdx.x * 256 + threadIdx.x;     // out index [l][k][o][ii]
  if (i >= 2L * 3 * CC2) return;
  long l  = i / (3 * CC2);
  long r  = i % (3 * CC2);
  long k  = r / CC2;
  long r2 = r % CC2;
  long o  = r2 / CH, ii = r2 % CH;
  out[i] = f2bf(cw[((l * CH + o) * CH + ii) * 3 + k]);
}

__global__ void cvt_bf16_kernel(const float* __restrict__ in, bf16_t* __restrict__ out, long n)
{
  long i = (long)blockIdx.x * 256 + threadIdx.x;
  if (i < n) out[i] = f2bf(in[i]);
}

// path [B][TX][TY] f32 -> pathT [B][TY][TX] bf16 (32x32 LDS tile transpose)
__global__ void transpose_bf16_kernel(const float* __restrict__ in, bf16_t* __restrict__ out)
{
  __shared__ float tile[32][33];
  const int b  = blockIdx.z;
  const int x0 = blockIdx.y * 32;   // TX
  const int y0 = blockIdx.x * 32;   // TY
  for (int i = threadIdx.y; i < 32; i += 8)
    tile[i][threadIdx.x] = in[((long)b * TXX + x0 + i) * TYY + y0 + threadIdx.x];
  __syncthreads();
  for (int i = threadIdx.y; i < 32; i += 8)
    out[((long)b * TYY + y0 + i) * TXX + x0 + threadIdx.x] = f2bf(tile[threadIdx.x][i]);
}

__global__ void add3_kernel(float* __restrict__ out, const float* __restrict__ p,
                            const float* __restrict__ e, long n)
{
  long i = (long)blockIdx.x * 256 + threadIdx.x;
  if (i < n) out[i] = out[i] + p[i] + e[i];
}

// ---------------------------------------------------------------------------
extern "C" void kernel_launch(void* const* d_in, const int* in_sizes, int n_in,
                              void* d_out, int out_size, void* d_ws, size_t ws_size,
                              hipStream_t stream)
{
  (void)in_sizes; (void)n_in; (void)out_size; (void)ws_size;
  const float* x      = (const float*)d_in[0];
  const float* x_mask = (const float*)d_in[1];
  const float* y_mask = (const float*)d_in[2];
  const float* pitch  = (const float*)d_in[3];
  const float* energy = (const float*)d_in[4];
  const float* path   = (const float*)d_in[5];
  const float* pr[3][6];  // per predictor: lng, lnb, cw, cb, ow, ob
  for (int p = 0; p < 3; ++p)
    for (int j = 0; j < 6; ++j)
      pr[p][j] = (const float*)d_in[6 + p * 6 + j];

  // workspace carve (256B aligned)
  char* ws = (char*)d_ws;
  size_t off = 0;
  auto carve = [&](size_t bytes) -> char* {
    char* p = ws + off;
    off = (off + bytes + 255) & ~(size_t)255;
    return p;
  };
  const long WSZ = 2L * 3 * CH * CH;                                  // per-predictor wbf elems
  bf16_t* wbf   = (bf16_t*)carve((size_t)3 * WSZ * sizeof(bf16_t));
  bf16_t* xbf   = (bf16_t*)carve((size_t)BB * CH * TXX * sizeof(bf16_t));
  bf16_t* pathT = (bf16_t*)carve((size_t)BB * TYY * TXX * sizeof(bf16_t));
  bf16_t* lnT   = (bf16_t*)carve((size_t)BB * TYY * CH * sizeof(bf16_t));
  float*  hbuf  = (float*) carve((size_t)BB * CH * TYY * sizeof(float));

  float* out_xe  = (float*)d_out;                          // [B][C][TY]
  float* out_dur = out_xe  + (size_t)BB * CH * TYY;        // [B][TX]
  float* out_pit = out_dur + (size_t)BB * TXX;             // [B][TY]
  float* out_ene = out_pit + (size_t)BB * TYY;             // [B][TY]

  // weight reorder + bf16 conversions
  for (int p = 0; p < 3; ++p)
    wprep_kernel<<<dim3((unsigned)((WSZ + 255) / 256)), 256, 0, stream>>>(
        pr[p][2], wbf + (size_t)p * WSZ);
  {
    long n = (long)BB * CH * TXX;
    cvt_bf16_kernel<<<dim3((unsigned)((n + 255) / 256)), 256, 0, stream>>>(x, xbf, n);
  }
  transpose_bf16_kernel<<<dim3(TYY / 32, TXX / 32, BB), dim3(32, 8), 0, stream>>>(path, pathT);

  // xe = bmm(x, path) -> written straight into d_out
  wmma_gemm_kernel<<<dim3(TYY / 64, CH / 128, BB), 256, 0, stream>>>(
      xbf, (long)CH * TXX, 0L, 1,
      pathT, (long)TYY * TXX,
      out_xe, (long)CH * TYY,
      TYY, TXX, nullptr, nullptr, 0);

  auto predictor = [&](const float* input, int T, const float* mask, int p, float* pred_out) {
    const bf16_t* wp = wbf + (size_t)p * WSZ;
    for (int l = 0; l < 2; ++l) {
      const float* src = (l == 0) ? input : hbuf;
      ln_to_bf16T_kernel<<<dim3(T / 32, BB), 256, 0, stream>>>(
          src, pr[p][0] + l * CH, pr[p][1] + l * CH, lnT, T);
      wmma_gemm_kernel<<<dim3(T / 64, CH / 128, BB), 256, 0, stream>>>(
          wp + (size_t)l * 3 * CH * CH, 0L, (long)CH * CH, 3,
          lnT, (long)T * CH,
          hbuf, (long)CH * T,
          T, CH, pr[p][3] + l * CH, mask, 1);
    }
    conv1x1_kernel<<<dim3(T / 32, BB), 256, 0, stream>>>(
        hbuf, pr[p][4], pr[p][5], mask, pred_out, T);
  };

  predictor(x,      TXX, x_mask, 0, out_dur);   // duration (reads x)
  predictor(out_xe, TYY, y_mask, 1, out_pit);   // pitch    (reads xe)
  predictor(out_xe, TYY, y_mask, 2, out_ene);   // energy   (reads xe)

  // out = xe + pitch + energy (in place, after predictors consumed xe)
  {
    long n = (long)BB * CH * TYY;
    add3_kernel<<<dim3((unsigned)((n + 255) / 256)), 256, 0, stream>>>(out_xe, pitch, energy, n);
  }
}
